// CosLocalDynamicsV2_88158498718221
// MI455X (gfx1250) — matrix operand order repaired and checked by reading, not compiled, not run on hardware
//
#include <hip/hip_runtime.h>
#include <hip/hip_bf16.h>

typedef __attribute__((ext_vector_type(16))) _Float16 v16h;
typedef __attribute__((ext_vector_type(8)))  float    v8f;
typedef __attribute__((ext_vector_type(4)))  float    v4f;

#define CH   256
#define NSP  4096        // 64*64
#define BB   8

#define SPITCH 264       // halfs per LDS row for sim staging (528B: 132 dw, %64 = 4)
#define HPITCH 520       // halfs per LDS row for conv staging (1040B: 260 dw, %64 = 4)

union FragAB {
    v16h h;
    v4f  q[2];
};

__device__ __forceinline__ void async_b128(unsigned lds_addr, const void* gaddr) {
    asm volatile("global_load_async_to_lds_b128 %0, %1, off"
                 :: "v"(lds_addr), "v"(gaddr) : "memory");
}
__device__ __forceinline__ void wait_async0() {
    asm volatile("s_wait_asynccnt 0x0" ::: "memory");
}

// ---------------------------------------------------------------------------
// Kernel 1: per-position L2 norms; write q_t, s_t (B,N,C) f16 and qscale (B,N)
// ---------------------------------------------------------------------------
__global__ __launch_bounds__(256) void prep_kernel(
    const float* __restrict__ x,      // (B,C,N)
    const float* __restrict__ maskN,  // (B,N)
    _Float16* __restrict__ qt,        // (B,N,C) f16
    _Float16* __restrict__ st,        // (B,N,C) f16
    float* __restrict__ qscale,       // (B,N)
    unsigned* __restrict__ fore0key)  // (B)
{
    int blk = blockIdx.x;
    int b   = blk >> 6;
    int n0  = (blk & 63) << 6;
    int tid = threadIdx.x;
    int sub = tid >> 6;
    int pos = tid & 63;
    int n   = n0 + pos;

    const float* xb = x + (size_t)b * CH * NSP;

    float ss = 0.f;
    for (int k = 0; k < 64; ++k) {
        float v = xb[(size_t)(sub * 64 + k) * NSP + n];
        ss += v * v;
    }
    __shared__ float red[256];
    red[tid] = ss;
    __syncthreads();
    float tot = red[pos] + red[64 + pos] + red[128 + pos] + red[192 + pos];

    float nrm = sqrtf(tot);
    float mv  = maskN[(size_t)b * NSP + n];            // exactly 0.0 or 1.0
    float qsc = mv / (mv * nrm + 1e-8f);
    float ssc = (1.0f - mv) / ((1.0f - mv) * nrm + 1e-8f);

    if (sub == 0) qscale[(size_t)b * NSP + n] = qsc;

    _Float16* qrow = qt + ((size_t)b * NSP + n) * CH + sub * 64;
    _Float16* srow = st + ((size_t)b * NSP + n) * CH + sub * 64;
    for (int k = 0; k < 64; ++k) {
        float v = xb[(size_t)(sub * 64 + k) * NSP + n];
        qrow[k] = (_Float16)(v * qsc);
        srow[k] = (_Float16)(v * ssc);
    }
    if (blk == 0 && tid < BB) fore0key[tid] = 0u;
}

// ---------------------------------------------------------------------------
// Kernel 2: Wc (C,2C) f32 -> f16
// ---------------------------------------------------------------------------
__global__ __launch_bounds__(256) void wc_half_kernel(
    const float* __restrict__ Wc, _Float16* __restrict__ Wh)
{
    int i = blockIdx.x * 256 + threadIdx.x;
    Wh[i] = (_Float16)Wc[i];
}

// ---------------------------------------------------------------------------
// Kernel 3: fused sim GEMM + per-row top-1 argmax + fore_wv[:,0] atomic-max.
// Workgroup = 128 queries (wave w owns rows n0+16w .. +15, A frags hoisted).
// All waves sweep 256 support tiles; each 8KB s-tile is staged once into LDS
// via async DMA (double-buffered) and consumed by all 8 waves.
// grid = B * 32, block = 256.
// ---------------------------------------------------------------------------
__global__ __launch_bounds__(256) void sim_argmax_kernel(
    const _Float16* __restrict__ qt,  // (B,N,C)
    const _Float16* __restrict__ st,  // (B,N,C)
    float* __restrict__ wv,           // (B,N)
    int*   __restrict__ idxb,         // (B,N)
    unsigned* __restrict__ fore0key)  // (B)
{
    int b    = blockIdx.x >> 5;
    int n0   = (blockIdx.x & 31) << 7;
    int tid  = threadIdx.x;
    int wave = tid >> 5;
    int lane = tid & 31;
    int lrow = lane & 15;
    int lhi  = lane >> 4;

    __align__(16) __shared__ _Float16 sbuf[2][16 * SPITCH];

    // hoist this wave's A fragments (16 query rows, full K=256) into registers
    const _Float16* qbase = qt + ((size_t)b * NSP + n0 + wave * 16 + lrow) * CH;
    FragAB afr[8];
#pragma unroll
    for (int k = 0; k < 8; ++k) {
        afr[k].q[0] = *(const v4f*)(qbase + k * 32 + (lhi << 3));
        afr[k].q[1] = *(const v4f*)(qbase + k * 32 + 16 + (lhi << 3));
    }

    const _Float16* sb = st + (size_t)b * NSP * CH;

    // stage support tile j (16 rows x 512B) into sbuf[p]; 2 async b128 / thread
    auto stage = [&](int j, int p) {
#pragma unroll
        for (int c = tid; c < 512; c += 256) {
            int row = c >> 5;
            int off = (c & 31) << 4;                         // bytes within row
            const char* g = (const char*)(sb + (size_t)(j * 16 + row) * CH) + off;
            unsigned l = (unsigned)(size_t)(&sbuf[p][row * SPITCH]) + off;
            async_b128(l, g);
        }
    };

    float runV[8];
    int   runI[8];
#pragma unroll
    for (int r = 0; r < 8; ++r) { runV[r] = -1e30f; runI[r] = 0; }

    stage(0, 0);

    for (int j = 0; j < 256; ++j) {
        int cur = j & 1;
        wait_async0();
        __syncthreads();
        if (j + 1 < 256) stage(j + 1, cur ^ 1);

        const _Float16* bbase = &sbuf[cur][lrow * SPITCH + (lhi << 3)];

        // load ALL B fragments first (in-order DS returns -> partial waits),
        // then run the WMMA accumulation chain.
        FragAB bf[8];
#pragma unroll
        for (int k = 0; k < 8; ++k) {
            bf[k].q[0] = *(const v4f*)(bbase + k * 32);
            bf[k].q[1] = *(const v4f*)(bbase + k * 32 + 16);
        }
        v8f acc = {};
#pragma unroll
        for (int k = 0; k < 8; ++k) {
            acc = __builtin_amdgcn_wmma_f32_16x16x32_f16(
                false, afr[k].h, false, bf[k].h, (short)0, acc, false, false);
        }

        int mycol = (j << 4) + lrow;     // strictly increasing per lane
#pragma unroll
        for (int r = 0; r < 8; ++r) {
            float v = acc[r];
            if (v > runV[r]) { runV[r] = v; runI[r] = mycol; }
        }

        if (j == 0) {   // column 0: lanes with lrow==0 (lanes 0 & 16), all waves
            float cm = acc[0];
#pragma unroll
            for (int r = 1; r < 8; ++r) cm = fmaxf(cm, acc[r]);
            float other = __shfl_xor(cm, 16, 32);
            cm = fmaxf(cm, other);
            if (lane == 0) {
                unsigned u = __float_as_uint(cm);
                unsigned key = (u & 0x80000000u) ? ~u : (u | 0x80000000u);
                atomicMax(&fore0key[b], key);
            }
        }
    }

    // reduce across the 16 lanes of each half (first-index tie-break)
#pragma unroll
    for (int m = 1; m < 16; m <<= 1) {
#pragma unroll
        for (int r = 0; r < 8; ++r) {
            float ov = __shfl_xor(runV[r], m, 32);
            int   oi = __shfl_xor(runI[r], m, 32);
            if (ov > runV[r] || (ov == runV[r] && oi < runI[r])) {
                runV[r] = ov; runI[r] = oi;
            }
        }
    }
    if (lrow == 0) {
        size_t base = (size_t)b * NSP + n0 + wave * 16 + (lhi << 3);
#pragma unroll
        for (int r = 0; r < 8; ++r) {
            wv[base + r]   = runV[r];
            idxb[base + r] = runI[r];
        }
    }
}

// ---------------------------------------------------------------------------
// Kernel 4: softmax over wv (per batch) + valid[b]
// ---------------------------------------------------------------------------
__global__ __launch_bounds__(256) void softmax_valid_kernel(
    const float* __restrict__ wv, const float* __restrict__ maskN,
    const unsigned* __restrict__ fore0key,
    float* __restrict__ w, float* __restrict__ validv)
{
    int b = blockIdx.x, tid = threadIdx.x;
    const float* row = wv + (size_t)b * NSP;
    __shared__ float red[256];

    float mx = -1e30f;
    for (int i = tid; i < NSP; i += 256) mx = fmaxf(mx, row[i]);
    red[tid] = mx; __syncthreads();
    for (int s = 128; s > 0; s >>= 1) {
        if (tid < s) red[tid] = fmaxf(red[tid], red[tid + s]);
        __syncthreads();
    }
    mx = red[0]; __syncthreads();

    float sum = 0.f;
    for (int i = tid; i < NSP; i += 256) sum += expf(row[i] - mx);
    red[tid] = sum; __syncthreads();
    for (int s = 128; s > 0; s >>= 1) {
        if (tid < s) red[tid] += red[tid + s];
        __syncthreads();
    }
    float inv = 1.0f / red[0];
    for (int i = tid; i < NSP; i += 256)
        w[(size_t)b * NSP + i] = expf(row[i] - mx) * inv;

    if (tid == 0) {
        unsigned k = fore0key[b];
        float f = (k & 0x80000000u) ? __uint_as_float(k & 0x7FFFFFFFu)
                                    : __uint_as_float(~k);
        validv[b] = (f > 0.5f) ? maskN[(size_t)b * NSP] : 0.0f;
    }
}

// ---------------------------------------------------------------------------
// Kernel 5: hybrid_t (B,N,2C) f16 = [ s_t[b,idx[b,n],:]*w[b,n] | q_t[b,n,:] ]
// ---------------------------------------------------------------------------
__global__ __launch_bounds__(256) void hybrid_kernel(
    const _Float16* __restrict__ qt, const _Float16* __restrict__ st,
    const int* __restrict__ idxb, const float* __restrict__ w,
    _Float16* __restrict__ hyb)
{
    size_t t   = (size_t)blockIdx.x * 256 + threadIdx.x;
    size_t row = t >> 6;
    int  part  = (int)(t & 63);
    int  b     = (int)(row >> 12);
    _Float16* dst = hyb + row * 512 + (size_t)part * 8;

    if (part < 32) {
        int   j   = idxb[row];
        float wgt = w[row];
        const _Float16* src = st + ((size_t)b * NSP + j) * CH + (size_t)part * 8;
#pragma unroll
        for (int k = 0; k < 8; ++k) dst[k] = (_Float16)((float)src[k] * wgt);
    } else {
        const _Float16* src = qt + row * CH + (size_t)(part - 32) * 8;
        *(v4f*)dst = *(const v4f*)src;
    }
}

// ---------------------------------------------------------------------------
// Kernel 6: 1x1-conv GEMM via WMMA + epilogue blend -> d_out (B,C,N)
// Workgroup = one 16-col hybrid tile (async-staged to LDS once, 8 waves
// consume it for 128 output channels). grid = B * 256 n-tiles * 2 o-halves.
// ---------------------------------------------------------------------------
__global__ __launch_bounds__(256) void conv_kernel(
    const _Float16* __restrict__ Wh,    // (256,512)
    const _Float16* __restrict__ hyb,   // (B,N,512)
    const float* __restrict__ bc,       // (256)
    const float* __restrict__ x,        // (B,C,N)
    const float* __restrict__ maskN,    // (B,N)
    const float* __restrict__ qscale,   // (B,N)
    const float* __restrict__ validv,   // (B)
    float* __restrict__ out)            // (B,C,N)
{
    int gb   = blockIdx.x;
    int b    = gb >> 9;
    int rem  = gb & 511;
    int nt   = rem >> 1;
    int half = rem & 1;
    int n0   = nt << 4;
    int tid  = threadIdx.x;
    int wave = tid >> 5;
    int lane = tid & 31;
    int lrow = lane & 15;
    int lhi  = lane >> 4;
    int o0   = (half << 7) + (wave << 4);

    __align__(16) __shared__ _Float16 hbuf[16 * HPITCH];

    // stage hybrid tile: 16 rows x 1024B, 4 async b128 per thread
    const _Float16* hb = hyb + ((size_t)b * NSP + n0) * 512;
#pragma unroll
    for (int c = tid; c < 1024; c += 256) {
        int row = c >> 6;
        int off = (c & 63) << 4;
        const char* g = (const char*)(hb + (size_t)row * 512) + off;
        unsigned l = (unsigned)(size_t)(&hbuf[row * HPITCH]) + off;
        async_b128(l, g);
    }
    wait_async0();
    __syncthreads();

    const _Float16* abase = Wh + (size_t)(o0 + lrow) * 512;
    const _Float16* bbase = hbuf + lrow * HPITCH + (lhi << 3);

    v8f acc = {};
    // two half-K groups of 8 fragments: batch loads, then WMMA chain
#pragma unroll
    for (int h = 0; h < 2; ++h) {
        FragAB a[8], bf[8];
#pragma unroll
        for (int k = 0; k < 8; ++k) {
            int k0 = h * 256 + k * 32;
            a[k].q[0]  = *(const v4f*)(abase + k0 + (lhi << 3));
            a[k].q[1]  = *(const v4f*)(abase + k0 + 16 + (lhi << 3));
            bf[k].q[0] = *(const v4f*)(bbase + k0);
            bf[k].q[1] = *(const v4f*)(bbase + k0 + 16);
        }
#pragma unroll
        for (int k = 0; k < 8; ++k) {
            acc = __builtin_amdgcn_wmma_f32_16x16x32_f16(
                false, a[k].h, false, bf[k].h, (short)0, acc, false, false);
        }
    }

    int   n  = n0 + lrow;
    float mv = maskN[(size_t)b * NSP + n];
    float vd = validv[b];
    float qs = qscale[(size_t)b * NSP + n];
#pragma unroll
    for (int r = 0; r < 8; ++r) {
        int o = o0 + r + (lhi << 3);
        size_t xi = ((size_t)b * CH + o) * NSP + n;
        float xv = x[xi];
        float res;
        if (mv == 0.0f)      res = xv;
        else if (vd != 0.0f) res = acc[r] + bc[o];
        else                 res = xv * qs;
        out[xi] = res;
    }
}

// ---------------------------------------------------------------------------
// Kernel 7: flags[j]=1 for j in idx[last batch]
// ---------------------------------------------------------------------------
__global__ __launch_bounds__(1024) void flags_kernel(
    const int* __restrict__ idxb, float* __restrict__ flags)
{
    int tid = threadIdx.x;
    for (int j = tid; j < NSP; j += 1024) flags[j] = 0.0f;
    __syncthreads();
    for (int j = tid; j < NSP; j += 1024)
        flags[idxb[(size_t)(BB - 1) * NSP + j]] = 1.0f;
}

// ---------------------------------------------------------------------------
// Kernel 8: attmap (B,1,512,512) = 8x nearest upsample of flags
// ---------------------------------------------------------------------------
__global__ __launch_bounds__(256) void attmap_kernel(
    const float* __restrict__ flags, float* __restrict__ att)
{
    int i  = blockIdx.x * 256 + threadIdx.x;
    int p  = i & (512 * 512 - 1);
    int hh = p >> 9, ww = p & 511;
    att[i] = flags[((hh >> 3) << 6) + (ww >> 3)];
}

// ---------------------------------------------------------------------------
extern "C" void kernel_launch(void* const* d_in, const int* in_sizes, int n_in,
                              void* d_out, int out_size, void* d_ws, size_t ws_size,
                              hipStream_t stream) {
    const float* x     = (const float*)d_in[0];   // (8,256,64,64)
    const float* maskN = (const float*)d_in[1];   // (8,1,64,64)
    const float* Wc    = (const float*)d_in[2];   // (256,512)
    const float* bc    = (const float*)d_in[3];   // (256)
    float* out = (float*)d_out;

    char* ws = (char*)d_ws;
    size_t off = 0;
    auto carve = [&](size_t bytes) { char* p = ws + off; off += (bytes + 255) & ~(size_t)255; return p; };
    _Float16* qt     = (_Float16*)carve((size_t)BB * NSP * CH * 2);
    _Float16* st     = (_Float16*)carve((size_t)BB * NSP * CH * 2);
    _Float16* hyb    = (_Float16*)carve((size_t)BB * NSP * 2 * CH * 2);
    _Float16* Wh     = (_Float16*)carve((size_t)CH * 2 * CH * 2);
    float*    qscale = (float*)carve((size_t)BB * NSP * 4);
    float*    wv     = (float*)carve((size_t)BB * NSP * 4);
    int*      idxb   = (int*)  carve((size_t)BB * NSP * 4);
    float*    wbuf   = (float*)carve((size_t)BB * NSP * 4);
    unsigned* fkey   = (unsigned*)carve(BB * 4);
    float*    validv = (float*)carve(BB * 4);
    float*    flags  = (float*)carve(NSP * 4);

    prep_kernel<<<BB * (NSP / 64), 256, 0, stream>>>(x, maskN, qt, st, qscale, fkey);
    wc_half_kernel<<<(CH * 2 * CH) / 256, 256, 0, stream>>>(Wc, Wh);
    sim_argmax_kernel<<<BB * 32, 256, 0, stream>>>(qt, st, wv, idxb, fkey);
    softmax_valid_kernel<<<BB, 256, 0, stream>>>(wv, maskN, fkey, wbuf, validv);
    hybrid_kernel<<<(int)(((size_t)BB * NSP * 512 / 8) / 256), 256, 0, stream>>>(qt, st, idxb, wbuf, hyb);
    conv_kernel<<<BB * 256 * 2, 256, 0, stream>>>(Wh, hyb, bc, x, maskN, qscale, validv, out);
    flags_kernel<<<1, 1024, 0, stream>>>(idxb, flags);
    attmap_kernel<<<(BB * 512 * 512) / 256, 256, 0, stream>>>(flags, out + (size_t)BB * CH * NSP);
}
